// AttentionOp_9019431322323
// MI455X (gfx1250) — compile-verified, matrix operand order checked by reading, not run on hardware
//
#include <hip/hip_runtime.h>

// CDNA5 / gfx1250 sliding-window causal flash-attention.
// Pass 1: convert Q (pre-scaled) and K to f16 in workspace.
// Pass 2: transpose V to [b,h,d,t] f16 in workspace.
// Pass 3: flash attention. 128-thread workgroup (4 waves) covers 64 Q rows and
//         shares double-buffered K/V tiles through LDS; v_wmma_f32_16x16x32_f16.

typedef __attribute__((ext_vector_type(16))) _Float16 v16h;
typedef __attribute__((ext_vector_type(8)))  _Float16 hv8;
typedef __attribute__((ext_vector_type(4)))  _Float16 hv4;
typedef __attribute__((ext_vector_type(8)))  float    v8f;

constexpr int kT = 2048;   // sequence length
constexpr int kH = 16;     // heads
constexpr int kD = 128;    // head dim
constexpr int BM = 16;     // query rows per wave
constexpr int BN = 32;     // keys per inner block
constexpr int NW = 4;      // waves per workgroup
constexpr int KPAD = 8;    // halves of padding (16B) -> conflict-free row-strided b128
constexpr int VPAD = 8;
constexpr float kScale = 0.08838834764831845f; // 1/sqrt(128)

static __device__ __forceinline__ int imin(int a, int b) { return a < b ? a : b; }
static __device__ __forceinline__ int imax(int a, int b) { return a > b ? a : b; }

static __device__ __forceinline__ float rowmax16(float x) {
#pragma unroll
  for (int off = 1; off < 16; off <<= 1) x = fmaxf(x, __shfl_xor(x, off, 32));
  return x;
}
static __device__ __forceinline__ float rowsum16(float x) {
#pragma unroll
  for (int off = 1; off < 16; off <<= 1) x += __shfl_xor(x, off, 32);
  return x;
}
static __device__ __forceinline__ v16h cat8(hv8 lo, hv8 hi) {
  return __builtin_shufflevector(lo, hi, 0, 1, 2, 3, 4, 5, 6, 7,
                                 8, 9, 10, 11, 12, 13, 14, 15);
}

// ---------------- Pass 1: Q (scaled) and K -> f16 --------------------------
__global__ void cvt_qk_kernel(const float* __restrict__ q, const float* __restrict__ k,
                              _Float16* __restrict__ qh, _Float16* __restrict__ kh,
                              long n4) {
  long i = (long)blockIdx.x * blockDim.x + threadIdx.x;
  if (i >= n4) return;
  float4 qv = ((const float4*)q)[i];
  float4 kv = ((const float4*)k)[i];
  hv4 qo, ko;
  qo[0] = (_Float16)(qv.x * kScale); qo[1] = (_Float16)(qv.y * kScale);
  qo[2] = (_Float16)(qv.z * kScale); qo[3] = (_Float16)(qv.w * kScale);
  ko[0] = (_Float16)kv.x; ko[1] = (_Float16)kv.y;
  ko[2] = (_Float16)kv.z; ko[3] = (_Float16)kv.w;
  ((hv4*)qh)[i] = qo;
  ((hv4*)kh)[i] = ko;
}

// ---------------- Pass 2: V [b,t,h,d] f32 -> Vt [b,h,d,t] f16 --------------
__global__ __launch_bounds__(256) void transpose_v_kernel(
    const float* __restrict__ v, _Float16* __restrict__ vt) {
  __shared__ float tile[16][17];
  const int t0 = blockIdx.x * 16, d0 = blockIdx.y * 16;
  const int h = blockIdx.z % kH, b = blockIdx.z / kH;
  const int tx = threadIdx.x, ty = threadIdx.y;
  tile[ty][tx] = v[(((size_t)b * kT + t0 + ty) * kH + h) * (size_t)kD + d0 + tx];
  __syncthreads();
  const int d = d0 + ty, t = t0 + tx;
  vt[(((size_t)b * kH + h) * kD + d) * (size_t)kT + t] = (_Float16)tile[tx][ty];
}

// ---------------- Pass 3: flash attention (f16, LDS-shared tiles) ----------
__global__ __launch_bounds__(NW * 32) void fa_swa_wmma_f16_kernel(
    const _Float16* __restrict__ qh, const _Float16* __restrict__ kh,
    const _Float16* __restrict__ vt, const int* __restrict__ wptr,
    float* __restrict__ out, int B) {
  const int W  = wptr[0];
  const int h = blockIdx.y, b = blockIdx.z;
  const int q0 = blockIdx.x * (BM * NW);          // 64 rows per workgroup
  const int tid  = threadIdx.x;
  const int wid  = tid >> 5;
  const int lane = tid & 31;
  const int ln = lane & 15;   // row (A) / col (B,C) index within 16
  const int hl = lane >> 4;   // lane half selector
  const int q0w = q0 + BM * wid;                  // this wave's 16 Q rows

  __shared__ _Float16 Klds[2][BN][kD + KPAD];     // keys x dims   (row-major)
  __shared__ _Float16 Vlds[2][kD][BN + VPAD];     // dims x keys   (from Vt)
  __shared__ _Float16 Plds[NW][BM][BN];           // per-wave P transpose

  const _Float16* khb = kh + ((size_t)b * kT * kH + h) * (size_t)kD;  // + key*kH*kD
  const _Float16* vtb = vt + ((size_t)b * kH + h) * kD * (size_t)kT;  // + d*kT

  // ---- Q tile -> A fragments --------------------------------------------
  v16h aq[4];
  {
    const _Float16* qrow = qh + ((((size_t)b * kT) + q0w + ln) * kH + h) * (size_t)kD;
#pragma unroll
    for (int c = 0; c < 4; ++c) {
      const int base = 32 * c + 8 * hl;
      aq[c] = cat8(*(const hv8*)(qrow + base), *(const hv8*)(qrow + base + 16));
    }
  }

  float m_run[8], l_run[8];
  v8f o[8] = {};
#pragma unroll
  for (int r = 0; r < 8; ++r) { m_run[r] = -1e30f; l_run[r] = 0.0f; }

  // block-uniform key range covering the union of the 4 waves' windows
  const int j_lo  = imax(0, q0 - W + 1);
  const int kb0   = (j_lo / BN) * BN;
  const int kb_hi = q0 + BM * NW - 1;

  // ---- cooperative tile staging (128 threads) ---------------------------
  auto stage = [&](int buf, int kbn) {
    {  // K tile: 32 keys x 128 halves; 64B per thread
      const int row = tid >> 2, seg = (tid & 3) * 32;
      const int key = imin(kbn + row, kT - 1);
      const hv8* src = (const hv8*)(khb + (size_t)key * (kH * kD) + seg);
      hv8* dst = (hv8*)&Klds[buf][row][seg];
#pragma unroll
      for (int i = 0; i < 4; ++i) dst[i] = src[i];
    }
    {  // Vt tile: 128 dims x 32 keys; 64B per thread (contiguous keys)
      const hv8* src = (const hv8*)(vtb + (size_t)tid * kT + kbn);
      hv8* dst = (hv8*)&Vlds[buf][tid][0];
#pragma unroll
      for (int i = 0; i < 4; ++i) dst[i] = src[i];
    }
  };

  stage(0, kb0);
  __syncthreads();

  int pbuf = 0;
  for (int kb = kb0; kb <= kb_hi; kb += BN) {
    if (kb + BN <= kb_hi) stage(1 - pbuf, kb + BN);  // prefetch next tile -> LDS

    // ---- S = Q K^T from LDS (8 wmma) -----------------------------------
    v8f s0 = {}, s1 = {};
#pragma unroll
    for (int c = 0; c < 4; ++c) {
      const int base = 32 * c + 8 * hl;
      v16h f0 = cat8(*(const hv8*)&Klds[pbuf][ln][base],
                     *(const hv8*)&Klds[pbuf][ln][base + 16]);
      v16h f1 = cat8(*(const hv8*)&Klds[pbuf][16 + ln][base],
                     *(const hv8*)&Klds[pbuf][16 + ln][base + 16]);
      s0 = __builtin_amdgcn_wmma_f32_16x16x32_f16(false, aq[c], false, f0,
                                                  (short)0, s0, false, false);
      s1 = __builtin_amdgcn_wmma_f32_16x16x32_f16(false, aq[c], false, f1,
                                                  (short)0, s1, false, false);
    }

    // ---- mask + online softmax (C layout: row = r + 8*hl, col = ln) ----
    const int col0 = kb + ln, col1 = col0 + 16;
    float p0[8], p1[8], alpha[8];
#pragma unroll
    for (int r = 0; r < 8; ++r) {
      const int i_abs = q0w + r + 8 * hl;
      const float x0 = ((col0 <= i_abs) && (col0 > i_abs - W)) ? s0[r] : -1e30f;
      const float x1 = ((col1 <= i_abs) && (col1 > i_abs - W)) ? s1[r] : -1e30f;
      const float mnew = fmaxf(m_run[r], rowmax16(fmaxf(x0, x1)));
      alpha[r] = __expf(m_run[r] - mnew);
      p0[r]    = __expf(x0 - mnew);
      p1[r]    = __expf(x1 - mnew);
      l_run[r] = l_run[r] * alpha[r] + rowsum16(p0[r] + p1[r]);
      m_run[r] = mnew;
    }

    // ---- P: C layout -> LDS -> A layout (per-wave region) --------------
#pragma unroll
    for (int r = 0; r < 8; ++r) {
      const int row = r + 8 * hl;
      Plds[wid][row][ln]      = (_Float16)p0[r];
      Plds[wid][row][ln + 16] = (_Float16)p1[r];
    }
    __syncthreads();
    v16h pa;
    {
      const int base = 8 * hl;
      pa = cat8(*(const hv8*)&Plds[wid][ln][base],
                *(const hv8*)&Plds[wid][ln][base + 16]);
    }

    // ---- rescale O, accumulate O += P * V (8 wmma) ---------------------
#pragma unroll
    for (int t = 0; t < 8; ++t) {
#pragma unroll
      for (int r = 0; r < 8; ++r) o[t][r] *= alpha[r];
    }
#pragma unroll
    for (int t = 0; t < 8; ++t) {
      const int dcol = 16 * t + ln;
      v16h bv = cat8(*(const hv8*)&Vlds[pbuf][dcol][8 * hl],
                     *(const hv8*)&Vlds[pbuf][dcol][8 * hl + 16]);
      o[t] = __builtin_amdgcn_wmma_f32_16x16x32_f16(false, pa, false, bv,
                                                    (short)0, o[t], false, false);
    }
    __syncthreads();   // tile consumed; next staging may overwrite
    pbuf ^= 1;
  }

  // ---- normalize and store ----------------------------------------------
#pragma unroll
  for (int r = 0; r < 8; ++r) {
    const int row = q0w + r + 8 * hl;
    const float inv = 1.0f / l_run[r];
    float* orow = out + ((((size_t)b * kT) + row) * kH + h) * (size_t)kD;
#pragma unroll
    for (int t = 0; t < 8; ++t) orow[16 * t + ln] = o[t][r] * inv;
  }
}

// ---------------- Fallback: monolithic fp32-input kernel -------------------
__global__ __launch_bounds__(32) void fa_swa_wmma_f32in_kernel(
    const float* __restrict__ q, const float* __restrict__ k,
    const float* __restrict__ v, const int* __restrict__ wptr,
    float* __restrict__ out, int B) {
  const int W  = wptr[0];
  const int qt = blockIdx.x, h = blockIdx.y, b = blockIdx.z;
  const int q0 = qt * BM;
  const int lane = threadIdx.x & 31;
  const int ln = lane & 15, hl = lane >> 4;

  __shared__ float    Vlds[BN][kD + 4];
  __shared__ _Float16 Plds[BM][BN];

  v16h aq[4];
  {
    const float* qrow = q + ((((size_t)b * kT) + q0 + ln) * kH + h) * (size_t)kD;
#pragma unroll
    for (int c = 0; c < 4; ++c) {
      const int base = 32 * c + 8 * hl;
#pragma unroll
      for (int i = 0; i < 8; ++i) {
        aq[c][i]     = (_Float16)(qrow[base + i] * kScale);
        aq[c][8 + i] = (_Float16)(qrow[base + 16 + i] * kScale);
      }
    }
  }
  float m_run[8], l_run[8];
  v8f o[8] = {};
#pragma unroll
  for (int r = 0; r < 8; ++r) { m_run[r] = -1e30f; l_run[r] = 0.0f; }

  const int j_lo = imax(0, q0 - W + 1);
  const int kb0 = (j_lo / BN) * BN, kb_hi = q0 + BM - 1;

  for (int kb = kb0; kb <= kb_hi; kb += BN) {
    {
      const int key = imin(kb + lane, kT - 1);
      const float4* src = (const float4*)(v + ((((size_t)b * kT) + key) * kH + h) * (size_t)kD);
      float4* dst = (float4*)&Vlds[lane][0];
#pragma unroll
      for (int i = 0; i < kD / 4; ++i) dst[i] = src[i];
    }
    v8f s0 = {}, s1 = {};
    {
      const int key0 = imin(kb + ln, kT - 1);
      const int key1 = imin(kb + 16 + ln, kT - 1);
      const float* kr0 = k + ((((size_t)b * kT) + key0) * kH + h) * (size_t)kD;
      const float* kr1 = k + ((((size_t)b * kT) + key1) * kH + h) * (size_t)kD;
#pragma unroll
      for (int c = 0; c < 4; ++c) {
        const int base = 32 * c + 8 * hl;
        v16h b0, b1;
#pragma unroll
        for (int i = 0; i < 8; ++i) {
          b0[i] = (_Float16)kr0[base + i];     b0[8 + i] = (_Float16)kr0[base + 16 + i];
          b1[i] = (_Float16)kr1[base + i];     b1[8 + i] = (_Float16)kr1[base + 16 + i];
        }
        s0 = __builtin_amdgcn_wmma_f32_16x16x32_f16(false, aq[c], false, b0, (short)0, s0, false, false);
        s1 = __builtin_amdgcn_wmma_f32_16x16x32_f16(false, aq[c], false, b1, (short)0, s1, false, false);
      }
    }
    const int col0 = kb + ln, col1 = col0 + 16;
    float p0[8], p1[8], alpha[8];
#pragma unroll
    for (int r = 0; r < 8; ++r) {
      const int i_abs = q0 + r + 8 * hl;
      const float x0 = ((col0 <= i_abs) && (col0 > i_abs - W)) ? s0[r] : -1e30f;
      const float x1 = ((col1 <= i_abs) && (col1 > i_abs - W)) ? s1[r] : -1e30f;
      const float mnew = fmaxf(m_run[r], rowmax16(fmaxf(x0, x1)));
      alpha[r] = __expf(m_run[r] - mnew);
      p0[r] = __expf(x0 - mnew);
      p1[r] = __expf(x1 - mnew);
      l_run[r] = l_run[r] * alpha[r] + rowsum16(p0[r] + p1[r]);
      m_run[r] = mnew;
    }
#pragma unroll
    for (int r = 0; r < 8; ++r) {
      const int row = r + 8 * hl;
      Plds[row][ln] = (_Float16)p0[r];
      Plds[row][ln + 16] = (_Float16)p1[r];
    }
    __syncthreads();
    v16h pa;
    {
      const int base = 8 * hl;
      pa = cat8(*(const hv8*)&Plds[ln][base], *(const hv8*)&Plds[ln][base + 16]);
    }
#pragma unroll
    for (int t = 0; t < 8; ++t) {
#pragma unroll
      for (int r = 0; r < 8; ++r) o[t][r] *= alpha[r];
    }
#pragma unroll
    for (int t = 0; t < 8; ++t) {
      const int dcol = 16 * t + ln;
      v16h bv;
#pragma unroll
      for (int i = 0; i < 8; ++i) {
        bv[i]     = (_Float16)Vlds[8 * hl + i][dcol];
        bv[8 + i] = (_Float16)Vlds[8 * hl + 16 + i][dcol];
      }
      o[t] = __builtin_amdgcn_wmma_f32_16x16x32_f16(false, pa, false, bv, (short)0, o[t], false, false);
    }
    __syncthreads();
  }
#pragma unroll
  for (int r = 0; r < 8; ++r) {
    const int row = q0 + r + 8 * hl;
    const float inv = 1.0f / l_run[r];
    float* orow = out + ((((size_t)b * kT) + row) * kH + h) * (size_t)kD;
#pragma unroll
    for (int t = 0; t < 8; ++t) orow[16 * t + ln] = o[t][r] * inv;
  }
}

extern "C" void kernel_launch(void* const* d_in, const int* in_sizes, int n_in,
                              void* d_out, int out_size, void* d_ws, size_t ws_size,
                              hipStream_t stream) {
  const float* q = (const float*)d_in[0];
  const float* k = (const float*)d_in[1];
  const float* v = (const float*)d_in[2];
  const int*   w = (const int*)d_in[3];
  float* out = (float*)d_out;
  const int B = in_sizes[0] / (kT * kH * kD);

  const size_t nElem = (size_t)B * kT * kH * kD;
  const size_t need  = 3 * nElem * sizeof(_Float16) + 256;  // Qh, Kh, Vt + slack

  if (ws_size >= need) {
    _Float16* qh = (_Float16*)d_ws;
    _Float16* kh = qh + nElem;
    _Float16* vt = kh + nElem;

    const long n4 = (long)(nElem / 4);
    cvt_qk_kernel<<<(n4 + 255) / 256, 256, 0, stream>>>(q, k, qh, kh, n4);

    dim3 tgrid(kT / 16, kD / 16, B * kH);
    transpose_v_kernel<<<tgrid, dim3(16, 16), 0, stream>>>(v, vt);

    dim3 grid(kT / (BM * NW), kH, B);
    fa_swa_wmma_f16_kernel<<<grid, NW * 32, 0, stream>>>(qh, kh, vt, w, out, B);
  } else {
    dim3 grid(kT / BM, kH, B);
    fa_swa_wmma_f32in_kernel<<<grid, 32, 0, stream>>>(q, k, v, w, out, B);
  }
}
